// SimpleMamba_8641474200042
// MI455X (gfx1250) — matrix operand order repaired and checked by reading, not compiled, not run on hardware
//
#include <hip/hip_runtime.h>

#define DIM     1024
#define TLEN    2048
#define NBATCH  8
#define MROWS   (NBATCH * TLEN)   // 16384
#define L_CHUNK 128
#define NCHUNK  (TLEN / L_CHUNK)  // 16
#define NWG_SC  64                // scan-phase workgroups (one 16-col slice each)

typedef __attribute__((ext_vector_type(16))) __bf16 v16bf;
typedef __attribute__((ext_vector_type(8)))  __bf16 v8bf;
typedef __attribute__((ext_vector_type(8)))  float  v8f;

__device__ __forceinline__ __bf16 f2bf(float f) {
  union { float f; unsigned u; } v; v.f = f;
  unsigned r = v.u + 0x7FFFu + ((v.u >> 16) & 1u);   // round-to-nearest-even
  unsigned short h = (unsigned short)(r >> 16);
  union { unsigned short s; __bf16 b; } o; o.s = h;
  return o.b;
}

__device__ __forceinline__ v16bf cat8(v8bf lo, v8bf hi) {
  return __builtin_shufflevector(lo, hi, 0,1,2,3,4,5,6,7,8,9,10,11,12,13,14,15);
}

__device__ __forceinline__ v8f wmma_bf16(v16bf a, v16bf b, v8f c) {
  return __builtin_amdgcn_wmma_f32_16x16x32_bf16(false, a, false, b, (short)0, c,
                                                 false, false);
}

__device__ __forceinline__ void storev(float* p, float v)  { *p = v; }
__device__ __forceinline__ void storev(__bf16* p, float v) { *p = f2bf(v); }

// Generation-counter device-wide barrier (all NWG workgroups co-resident).
__device__ __forceinline__ void global_barrier(unsigned* cnt, unsigned* gen,
                                               unsigned round, unsigned nwg,
                                               int tid) {
  __threadfence();
  if (tid == 0) {
    unsigned arrived = atomicAdd(cnt, 1u);
    if (arrived == nwg - 1u) {
      atomicExch(cnt, 0u);
      __threadfence();
      atomicAdd(gen, 1u);
    }
    while (atomicAdd(gen, 0u) < round) __builtin_amdgcn_s_sleep(1);
  }
  __syncthreads();
}

// ---------------------------------------------------------------- cast kernel
__global__ void cast_f32_bf16_kernel(const float* __restrict__ in,
                                     __bf16* __restrict__ out, int n) {
  int i = blockIdx.x * blockDim.x + threadIdx.x;
  if (i < n) out[i] = f2bf(in[i]);
}

__global__ void init_bar_kernel(unsigned* bar) { bar[threadIdx.x] = 0u; }

// --------------------------------------------------------------- WMMA GEMM
// Y[M,1024] = X[M,1024] (f32|bf16) @ W[1024,1024] (bf16). Y f32 or bf16.
// WG tile 128x64, 8 waves -> 32x32 each (2x2 WMMA tiles). M = gridDim.x*128.
template <typename XT, typename YT>
__global__ __launch_bounds__(256)
void gemm_wmma_kernel(const XT* __restrict__ X, const __bf16* __restrict__ W,
                      YT* __restrict__ Y) {
  constexpr int LDA = 40;  // padded K-stride (bf16 elems), 16B aligned
  __shared__ __align__(16) __bf16 At[128 * LDA];   // [m][k]
  __shared__ __align__(16) __bf16 Bt[64 * LDA];    // transposed: [n][k]

  const int m0 = blockIdx.x * 128, n0 = blockIdx.y * 64;
  const int tid = threadIdx.x, lane = tid & 31, wave = tid >> 5;
  const int mw = wave & 3, nw = wave >> 2;

  v8f acc[2][2] = {};
  const int ar = lane & 15, akoff = (lane >> 4) * 8;
  const int bn = lane & 15, bkoff = (lane >> 4) * 8;

  for (int k0 = 0; k0 < DIM; k0 += 32) {
    __syncthreads();
    {   // stage A: 128 rows x 32 K (convert to bf16)
      int row = tid >> 1, ks = (tid & 1) * 16;
      const XT* src = X + (size_t)(m0 + row) * DIM + k0 + ks;
      __bf16* dst = &At[row * LDA + ks];
      #pragma unroll
      for (int i = 0; i < 16; i++) dst[i] = f2bf((float)src[i]);
    }
    {   // stage B transposed: 32 K-rows x 64 N
      int kk = tid >> 3, ns = (tid & 7) * 8;
      const __bf16* src = W + (size_t)(k0 + kk) * DIM + n0 + ns;
      #pragma unroll
      for (int i = 0; i < 8; i++) Bt[(ns + i) * LDA + kk] = src[i];
    }
    __syncthreads();

    v16bf a[2], b[2];
    #pragma unroll
    for (int tm = 0; tm < 2; tm++) {
      const __bf16* ap = &At[(mw * 32 + tm * 16 + ar) * LDA + akoff];
      a[tm] = cat8(*(const v8bf*)&ap[0], *(const v8bf*)&ap[16]);
    }
    #pragma unroll
    for (int tn = 0; tn < 2; tn++) {
      const __bf16* bp = &Bt[(nw * 32 + tn * 16 + bn) * LDA + bkoff];
      b[tn] = cat8(*(const v8bf*)&bp[0], *(const v8bf*)&bp[16]);
    }
    #pragma unroll
    for (int tm = 0; tm < 2; tm++)
      #pragma unroll
      for (int tn = 0; tn < 2; tn++)
        acc[tm][tn] = wmma_bf16(a[tm], b[tn], acc[tm][tn]);
  }

  const int nb = lane & 15, mh = (lane >> 4) * 8;
  #pragma unroll
  for (int tm = 0; tm < 2; tm++)
    #pragma unroll
    for (int tn = 0; tn < 2; tn++) {
      int ncol = n0 + nw * 32 + tn * 16 + nb;
      int mrow = m0 + mw * 32 + tm * 16 + mh;
      #pragma unroll
      for (int v = 0; v < 8; v++)
        storev(&Y[(size_t)(mrow + v) * DIM + ncol], acc[tm][tn][v]);
    }
}

// --------------------------------------------------- Phase 1: local scans
// All 16 chunks in parallel: state rows r = chunk*8 + batch (M=128, no pad).
// 64 WGs x 256 thr; WG owns 16 cols (A-slice LDS-resident); wave w owns
// M-tile w (16 rows) for the WG's N-tile; K=1024 -> 32 WMMAs per step.
__global__ __launch_bounds__(256)
void local_scan_kernel(const float* __restrict__ xB,
                       const __bf16* __restrict__ Abf,
                       __bf16* __restrict__ hbuf, unsigned* __restrict__ bar) {
  constexpr int LDH = DIM + 8;
  __shared__ __align__(16) __bf16 at[16 * LDH];    // A[:,n0:n0+16]^T
  const int tid = threadIdx.x, lane = tid & 31, wave = tid >> 5;
  const int n0 = blockIdx.x * 16;

  {   // stage A slice transposed (once)
    int kbase = tid * 4;
    for (int kk = 0; kk < 4; kk++) {
      int k = kbase + kk;
      const __bf16* src = Abf + (size_t)k * DIM + n0;
      #pragma unroll
      for (int n = 0; n < 16; n++) at[n * LDH + k] = src[n];
    }
  }
  __syncthreads();

  unsigned* cnt = bar; unsigned* gen = bar + 32;

  const int akoff = (lane >> 4) * 8;
  const int r_a = wave * 16 + (lane & 15);         // A-frag state row
  const size_t grow_a = (size_t)(r_a & 7) * TLEN + (size_t)(r_a >> 3) * L_CHUNK;
  const int bn = lane & 15, bkoff = (lane >> 4) * 8;
  const int cn = lane & 15, cmh = (lane >> 4) * 8;

  // s = 0: hloc_0 = xB_0 (per chunk)
  for (int e = tid; e < 128 * 16; e += 256) {
    int r = e >> 4, n = e & 15;
    size_t idx = ((size_t)(r & 7) * TLEN + (size_t)(r >> 3) * L_CHUNK) * DIM + n0 + n;
    hbuf[idx] = f2bf(xB[idx]);
  }
  global_barrier(cnt, gen, 1u, NWG_SC, tid);

  for (int s = 1; s < L_CHUNK; s++) {
    const __bf16* aprev = hbuf + (grow_a + (size_t)(s - 1)) * DIM;
    v8f acc = {};
    #pragma unroll 4
    for (int kb = 0; kb < 32; kb++) {
      int k = kb * 32;
      v16bf a = cat8(*(const v8bf*)&aprev[k + akoff],
                     *(const v8bf*)&aprev[k + akoff + 16]);
      const __bf16* bp = &at[bn * LDH + k + bkoff];
      v16bf b = cat8(*(const v8bf*)&bp[0], *(const v8bf*)&bp[16]);
      acc = wmma_bf16(a, b, acc);
    }
    #pragma unroll
    for (int v = 0; v < 8; v++) {
      int r = wave * 16 + cmh + v;
      size_t idx = ((size_t)(r & 7) * TLEN + (size_t)(r >> 3) * L_CHUNK + s) * DIM
                 + n0 + cn;
      hbuf[idx] = f2bf(acc[v] + xB[idx]);
    }
    global_barrier(cnt, gen, (unsigned)(s + 1), NWG_SC, tid);
  }
}

// --------------------------------------------------- Phase 2b: carry pass
// carry_c = hloc_end_c + carry_{c-1} @ A^L, c = 1..15 (15 serial rounds).
// 64 WGs x 128 thr (4 waves K-split), M=8 padded to 16 via shared zero row.
__global__ __launch_bounds__(128)
void carry_kernel(const __bf16* __restrict__ hbuf, const __bf16* __restrict__ AL,
                  __bf16* __restrict__ carr, unsigned* __restrict__ bar) {
  constexpr int LDH = DIM + 8;
  __shared__ __align__(16) __bf16 hlds[9 * LDH];   // rows 0..7 carry, row 8 zero
  __shared__ __align__(16) __bf16 pt[16 * LDH];    // (A^L)[:,n0:n0+16]^T
  __shared__ float partial[4][16][16];

  const int tid = threadIdx.x, lane = tid & 31, wave = tid >> 5;
  const int n0 = blockIdx.x * 16;

  for (int i = tid; i < 9 * LDH; i += 128) hlds[i] = (__bf16)0.0f;
  {   // stage A^L slice transposed
    int kbase = tid * 8;
    for (int kk = 0; kk < 8; kk++) {
      int k = kbase + kk;
      const __bf16* src = AL + (size_t)k * DIM + n0;
      #pragma unroll
      for (int n = 0; n < 16; n++) pt[n * LDH + k] = src[n];
    }
  }
  __syncthreads();

  unsigned* cnt = bar; unsigned* gen = bar + 32;
  const int ar = lane & 15;
  const int arow = (ar < 8) ? ar : 8;
  const int akoff = (lane >> 4) * 8;
  const int bn = lane & 15, bkoff = (lane >> 4) * 8;

  {   // carry_0 = hloc_end_0
    int b = tid >> 4, n = tid & 15;   // 128 threads = 8x16
    carr[(size_t)b * DIM + n0 + n] =
        hbuf[((size_t)b * TLEN + L_CHUNK - 1) * DIM + n0 + n];
  }
  global_barrier(cnt, gen, 1u, NWG_SC, tid);

  for (int c = 1; c < NCHUNK; c++) {
    {   // stage carry_{c-1} rows 0..7
      int m = tid >> 4, ks = (tid & 15) * 64;
      const uint4* s4 = (const uint4*)(carr + ((size_t)(c - 1) * 8 + m) * DIM + ks);
      uint4* d4 = (uint4*)&hlds[m * LDH + ks];
      #pragma unroll
      for (int i = 0; i < 8; i++) d4[i] = s4[i];
    }
    __syncthreads();

    v8f acc = {};
    #pragma unroll
    for (int kb = 0; kb < 8; kb++) {
      int k = wave * 256 + kb * 32;
      const __bf16* ap = &hlds[arow * LDH + k + akoff];
      v16bf a = cat8(*(const v8bf*)&ap[0], *(const v8bf*)&ap[16]);
      const __bf16* bp = &pt[bn * LDH + k + bkoff];
      v16bf b = cat8(*(const v8bf*)&bp[0], *(const v8bf*)&bp[16]);
      acc = wmma_bf16(a, b, acc);
    }
    {
      int n = lane & 15, mb = (lane >> 4) * 8;
      #pragma unroll
      for (int v = 0; v < 8; v++) partial[wave][mb + v][n] = acc[v];
    }
    __syncthreads();
    {
      int m = tid >> 4, n = tid & 15;  // 8x16 live elements
      float s = partial[0][m][n] + partial[1][m][n] +
                partial[2][m][n] + partial[3][m][n] +
                (float)hbuf[((size_t)m * TLEN + (size_t)c * L_CHUNK + L_CHUNK - 1)
                            * DIM + n0 + n];
      carr[((size_t)c * 8 + m) * DIM + n0 + n] = f2bf(s);
    }
    __syncthreads();
    global_barrier(cnt, gen, (unsigned)(c + 1), NWG_SC, tid);
  }
}

// --------------------------------------------------- Phase 3: fixup scans
// f_s = f_{s-1} @ A (f_{-1} = carry_{c-1}); h_{cL+s} += f_s, chunks 1..15 in
// parallel (rows r = (c-1)*8+b, 120 live + 8 zero-pad). 128 serial rounds.
__global__ __launch_bounds__(256)
void fixup_kernel(const __bf16* __restrict__ Abf, const __bf16* __restrict__ carr,
                  __bf16* __restrict__ hbuf, __bf16* __restrict__ f0,
                  __bf16* __restrict__ f1, unsigned* __restrict__ bar) {
  constexpr int LDH = DIM + 8;
  __shared__ __align__(16) __bf16 at[16 * LDH];
  const int tid = threadIdx.x, lane = tid & 31, wave = tid >> 5;
  const int n0 = blockIdx.x * 16;

  {
    int kbase = tid * 4;
    for (int kk = 0; kk < 4; kk++) {
      int k = kbase + kk;
      const __bf16* src = Abf + (size_t)k * DIM + n0;
      #pragma unroll
      for (int n = 0; n < 16; n++) at[n * LDH + k] = src[n];
    }
  }
  __syncthreads();

  unsigned* cnt = bar; unsigned* gen = bar + 32;
  const int akoff = (lane >> 4) * 8;
  const int r_a = wave * 16 + (lane & 15);
  const int bn = lane & 15, bkoff = (lane >> 4) * 8;
  const int cn = lane & 15, cmh = (lane >> 4) * 8;

  // f_prev := carry rows (carr row index == r for r < 120), pad rows zero
  for (int e = tid; e < 128 * 16; e += 256) {
    int r = e >> 4, n = e & 15;
    __bf16 v = (r < 120) ? carr[(size_t)r * DIM + n0 + n] : (__bf16)0.0f;
    f0[(size_t)r * DIM + n0 + n] = v;
  }
  global_barrier(cnt, gen, 1u, NWG_SC, tid);

  for (int s = 0; s < L_CHUNK; s++) {
    const __bf16* fprev = (s & 1) ? f1 : f0;
    __bf16*       fnext = (s & 1) ? f0 : f1;
    const __bf16* ap = fprev + (size_t)r_a * DIM;
    v8f acc = {};
    #pragma unroll 4
    for (int kb = 0; kb < 32; kb++) {
      int k = kb * 32;
      v16bf a = cat8(*(const v8bf*)&ap[k + akoff], *(const v8bf*)&ap[k + akoff + 16]);
      const __bf16* bp = &at[bn * LDH + k + bkoff];
      v16bf b = cat8(*(const v8bf*)&bp[0], *(const v8bf*)&bp[16]);
      acc = wmma_bf16(a, b, acc);
    }
    #pragma unroll
    for (int v = 0; v < 8; v++) {
      int r = wave * 16 + cmh + v;
      int n = n0 + cn;
      fnext[(size_t)r * DIM + n] = f2bf(acc[v]);
      if (r < 120) {
        int c = (r >> 3) + 1, b = r & 7;
        size_t idx = ((size_t)b * TLEN + (size_t)c * L_CHUNK + s) * DIM + n;
        hbuf[idx] = f2bf((float)hbuf[idx] + acc[v]);
      }
    }
    global_barrier(cnt, gen, (unsigned)(s + 2), NWG_SC, tid);
  }
}

// ---------------------------------------------------------------- launch
extern "C" void kernel_launch(void* const* d_in, const int* in_sizes, int n_in,
                              void* d_out, int out_size, void* d_ws, size_t ws_size,
                              hipStream_t stream) {
  const float* x = (const float*)d_in[0];
  const float* A = (const float*)d_in[1];
  const float* B = (const float*)d_in[2];
  const float* C = (const float*)d_in[3];
  float* out = (float*)d_out;

  char* ws = (char*)d_ws;
  size_t off = 0;
  auto alloc = [&](size_t bytes) -> void* {
    void* p = (void*)(ws + off);
    off += (bytes + 255) & ~(size_t)255;
    return p;
  };
  __bf16*   Abf  = (__bf16*)alloc((size_t)DIM * DIM * sizeof(__bf16));
  __bf16*   Bbf  = (__bf16*)alloc((size_t)DIM * DIM * sizeof(__bf16));
  __bf16*   Cbf  = (__bf16*)alloc((size_t)DIM * DIM * sizeof(__bf16));
  __bf16*   Pa   = (__bf16*)alloc((size_t)DIM * DIM * sizeof(__bf16));  // powers
  __bf16*   Pb   = (__bf16*)alloc((size_t)DIM * DIM * sizeof(__bf16));
  float*    xBuf = (float*) alloc((size_t)MROWS * DIM * sizeof(float));
  __bf16*   hbuf = (__bf16*)alloc((size_t)MROWS * DIM * sizeof(__bf16));
  __bf16*   carr = (__bf16*)alloc((size_t)NCHUNK * 8 * DIM * sizeof(__bf16));
  __bf16*   f0   = (__bf16*)alloc((size_t)128 * DIM * sizeof(__bf16));
  __bf16*   f1   = (__bf16*)alloc((size_t)128 * DIM * sizeof(__bf16));
  unsigned* bar  = (unsigned*)alloc(192 * sizeof(unsigned));

  const int nmat = DIM * DIM;
  cast_f32_bf16_kernel<<<(nmat + 255) / 256, 256, 0, stream>>>(A, Abf, nmat);
  cast_f32_bf16_kernel<<<(nmat + 255) / 256, 256, 0, stream>>>(B, Bbf, nmat);
  cast_f32_bf16_kernel<<<(nmat + 255) / 256, 256, 0, stream>>>(C, Cbf, nmat);
  init_bar_kernel<<<1, 192, 0, stream>>>(bar);

  // xB = x @ B
  gemm_wmma_kernel<float, float>
      <<<dim3(MROWS / 128, DIM / 64), 256, 0, stream>>>(x, Bbf, xBuf);

  // Phase 1: 16 parallel local scans (128 barrier rounds)
  local_scan_kernel<<<NWG_SC, 256, 0, stream>>>(xBuf, Abf, hbuf, bar);

  // Phase 2a: A^128 by 7 log-squarings (bf16-out WMMA GEMMs, M=1024)
  dim3 pg(DIM / 128, DIM / 64);
  gemm_wmma_kernel<__bf16, __bf16><<<pg, 256, 0, stream>>>(Abf, Abf, Pa); // A^2
  gemm_wmma_kernel<__bf16, __bf16><<<pg, 256, 0, stream>>>(Pa, Pa, Pb);  // A^4
  gemm_wmma_kernel<__bf16, __bf16><<<pg, 256, 0, stream>>>(Pb, Pb, Pa);  // A^8
  gemm_wmma_kernel<__bf16, __bf16><<<pg, 256, 0, stream>>>(Pa, Pa, Pb);  // A^16
  gemm_wmma_kernel<__bf16, __bf16><<<pg, 256, 0, stream>>>(Pb, Pb, Pa);  // A^32
  gemm_wmma_kernel<__bf16, __bf16><<<pg, 256, 0, stream>>>(Pa, Pa, Pb);  // A^64
  gemm_wmma_kernel<__bf16, __bf16><<<pg, 256, 0, stream>>>(Pb, Pb, Pa);  // A^128

  // Phase 2b: serial carry pass (15 barrier rounds)
  carry_kernel<<<NWG_SC, 128, 0, stream>>>(hbuf, Pa, carr, bar + 64);

  // Phase 3: parallel fixup scans (128 barrier rounds)
  fixup_kernel<<<NWG_SC, 256, 0, stream>>>(Abf, carr, hbuf, f0, f1, bar + 128);

  // out = h @ C
  gemm_wmma_kernel<__bf16, float>
      <<<dim3(MROWS / 128, DIM / 64), 256, 0, stream>>>(hbuf, Cbf, out);
}
